// GCN_54039278519097
// MI455X (gfx1250) — matrix operand order repaired and checked by reading, not compiled, not run on hardware
//
#include <hip/hip_runtime.h>
#include <hip/hip_bf16.h>
#include <math.h>

typedef __attribute__((ext_vector_type(16))) _Float16 v16h;
typedef __attribute__((ext_vector_type(8)))  _Float16 v8h;
typedef __attribute__((ext_vector_type(8)))  float    v8f;

#define HDIM 128

// ---------------- utility kernels ----------------

__global__ void gcn_zero_f32(float* __restrict__ p, long n) {
    long i = (long)blockIdx.x * blockDim.x + threadIdx.x;
    if (i < n) p[i] = 0.0f;
}

__global__ void gcn_degree(const int* __restrict__ src, const int* __restrict__ dst,
                           float* __restrict__ od, float* __restrict__ id, int E) {
    int e = blockIdx.x * blockDim.x + threadIdx.x;
    if (e < E) {
        atomicAdd(od + src[e], 1.0f);
        atomicAdd(id + dst[e], 1.0f);
    }
}

// norm = rsqrt(deg + 1): the +1 is the self-loop; deg+1 >= 1 so max() is implicit.
__global__ void gcn_norm_finalize(float* __restrict__ od, float* __restrict__ id, int N) {
    int i = blockIdx.x * blockDim.x + threadIdx.x;
    if (i < N) {
        od[i] = rsqrtf(od[i] + 1.0f);
        id[i] = rsqrtf(id[i] + 1.0f);
    }
}

// Wt[n*128 + k] = (f16) W[k*128 + n]  (transpose so B fragments load contiguously)
__global__ void gcn_wt_convert(const float* __restrict__ W, _Float16* __restrict__ Wt) {
    int i = blockIdx.x * blockDim.x + threadIdx.x;
    if (i < HDIM * HDIM) {
        int n = i >> 7, k = i & 127;
        Wt[i] = (_Float16)W[k * HDIM + n];
    }
}

// hs = f16(h * norm_src[row])
__global__ void gcn_scale_f16(const float* __restrict__ h, const float* __restrict__ ns,
                              _Float16* __restrict__ hs, long total) {
    long i = (long)blockIdx.x * blockDim.x + threadIdx.x;
    if (i < total) hs[i] = (_Float16)(h[i] * ns[i >> 7]);
}

// ---------------- WMMA GEMM: t[N,128] = hs[N,128] @ W[128,128] ----------------
// 8 waves/block, each wave computes a 16-row x 128-col panel with
// v_wmma_f32_16x16x32_f16 (8 N-tiles x 4 K-steps = 32 WMMA per wave).
// Tail handling: A-loads clamp the row; C-stores use one wave-uniform branch
// (full panels take the unguarded straight-line store path, no exec churn).

__global__ __launch_bounds__(256)
void gcn_gemm_wmma(const _Float16* __restrict__ hs, const _Float16* __restrict__ Wt,
                   float* __restrict__ t, int N) {
    const int lane = threadIdx.x & 31;
    const int wave = threadIdx.x >> 5;
    const int rowbase = blockIdx.x * 128 + wave * 16;
    if (rowbase >= N) return;                 // wave-uniform: EXEC stays all-1s

    const int m  = lane & 15;
    const int hi = lane >> 4;

    // --- A fragments: lane holds K = {kb+16*hi .. +7} and
    // {kb+16*hi+8 .. +7} as the low/high 8 halves of the 16-half fragment. ---
    int rg = rowbase + m;
    if (rg >= N) rg = N - 1;                  // clamp tail reads; stores are guarded
    const _Float16* arow = hs + (size_t)rg * HDIM;

    v16h afrag[4];
#pragma unroll
    for (int kt = 0; kt < 4; ++kt) {
        const int kb = kt * 32;
        v8h lo  = *(const v8h*)(arow + kb + 16 * hi);
        v8h hi8 = *(const v8h*)(arow + kb + 16 * hi + 8);
        afrag[kt] = __builtin_shufflevector(lo, hi8, 0, 1, 2, 3, 4, 5, 6, 7,
                                                     8, 9, 10, 11, 12, 13, 14, 15);
    }

    const int n = lane & 15;
    const bool full = (rowbase + 16 <= N);    // wave-uniform
    // lane's store base: row (rowbase + 8*hi), column n; row v adds v*HDIM
    float* crow = t + ((size_t)(rowbase + 8 * hi)) * HDIM + n;

#pragma unroll
    for (int nt = 0; nt < 8; ++nt) {
        const _Float16* brow = Wt + (size_t)(nt * 16 + n) * HDIM;
        v8f acc = {};
#pragma unroll
        for (int kt = 0; kt < 4; ++kt) {
            // B layout: VGPRs hold K = 16*hi .. 16*hi+15 contiguously for column n.
            v16h bfrag = *(const v16h*)(brow + kt * 32 + 16 * hi);
            acc = __builtin_amdgcn_wmma_f32_16x16x32_f16(
                false, afrag[kt], false, bfrag, (short)0, acc, false, false);
        }
        // C layout: lane -> column n, VGPR v -> row (v + 8*hi)
        if (full) {
#pragma unroll
            for (int v = 0; v < 8; ++v)
                crow[(size_t)v * HDIM + nt * 16] = acc[v];
        } else {
#pragma unroll
            for (int v = 0; v < 8; ++v) {
                int rr = rowbase + v + 8 * hi;
                if (rr < N) t[(size_t)rr * HDIM + nt * 16 + n] = acc[v];
            }
        }
    }
}

// ---------------- edge scatter-add (L2-resident atomics) ----------------
// One wave per edge; lane covers 4 contiguous floats. e in [E, E+N) = self loop.

__global__ void gcn_scatter(const int* __restrict__ src, const int* __restrict__ dst,
                            const float* __restrict__ t, float* __restrict__ agg,
                            int E, int N) {
    long idx = (long)blockIdx.x * blockDim.x + threadIdx.x;
    long e = idx >> 5;
    int lane = idx & 31;
    if (e >= (long)E + N) return;
    int s, d;
    if (e < E) { s = src[e]; d = dst[e]; }
    else       { s = d = (int)(e - E); }
    const float4 v = *(const float4*)(t + (size_t)s * HDIM + lane * 4);
    float* a = agg + (size_t)d * HDIM + lane * 4;
    atomicAdd(a + 0, v.x);
    atomicAdd(a + 1, v.y);
    atomicAdd(a + 2, v.z);
    atomicAdd(a + 3, v.w);
}

// h = relu(agg * norm_dst[row] + b[col])
__global__ void gcn_bias_relu(const float* __restrict__ agg, const float* __restrict__ nd,
                              const float* __restrict__ b, float* __restrict__ h, long total) {
    long i = (long)blockIdx.x * blockDim.x + threadIdx.x;
    if (i < total) {
        float v = agg[i] * nd[i >> 7] + b[i & 127];
        h[i] = v > 0.0f ? v : 0.0f;
    }
}

// ---------------- pooling: sigmoid-gated sum + max per graph ----------------
// One wave per node. h >= 0 post-ReLU, so float-max == int-max on bit patterns.

__global__ void gcn_pool(const float* __restrict__ h, const int* __restrict__ gid,
                         const float* __restrict__ Wp, const float* __restrict__ bp,
                         float* __restrict__ out, int N) {
    long idx = (long)blockIdx.x * blockDim.x + threadIdx.x;
    long node = idx >> 5;
    int lane = idx & 31;
    if (node >= N) return;
    const float4 v  = *(const float4*)(h + (size_t)node * HDIM + lane * 4);
    const float4 w4 = *(const float4*)(Wp + lane * 4);
    float p = v.x * w4.x + v.y * w4.y + v.z * w4.z + v.w * w4.w;
#pragma unroll
    for (int off = 16; off; off >>= 1) p += __shfl_xor(p, off, 32);
    float wgt = 1.0f / (1.0f + __expf(-(p + bp[0])));
    int g = gid[node];
    float* os = out + (size_t)g * 256 + lane * 4;   // g_sum: cols 0..127
    float* om = os + HDIM;                           // g_max: cols 128..255
    atomicAdd(os + 0, v.x * wgt);
    atomicAdd(os + 1, v.y * wgt);
    atomicAdd(os + 2, v.z * wgt);
    atomicAdd(os + 3, v.w * wgt);
    atomicMax((int*)(om + 0), __float_as_int(v.x));
    atomicMax((int*)(om + 1), __float_as_int(v.y));
    atomicMax((int*)(om + 2), __float_as_int(v.z));
    atomicMax((int*)(om + 3), __float_as_int(v.w));
}

// ---------------- launch ----------------

static inline int cdiv(long a, int b) { return (int)((a + b - 1) / b); }

extern "C" void kernel_launch(void* const* d_in, const int* in_sizes, int n_in,
                              void* d_out, int out_size, void* d_ws, size_t ws_size,
                              hipStream_t stream) {
    const float* node_feats = (const float*)d_in[0];
    const int*   src        = (const int*)d_in[1];
    const int*   dst        = (const int*)d_in[2];
    const int*   gid        = (const int*)d_in[3];
    const float* Ws[3]      = {(const float*)d_in[4], (const float*)d_in[6], (const float*)d_in[8]};
    const float* bs[3]      = {(const float*)d_in[5], (const float*)d_in[7], (const float*)d_in[9]};
    const float* W_pool     = (const float*)d_in[10];
    const float* b_pool     = (const float*)d_in[11];

    const int N = in_sizes[0] / HDIM;
    const int E = in_sizes[1];
    const long NH = (long)N * HDIM;

    // workspace partition (floats)
    float* ws       = (float*)d_ws;
    float* norm_src = ws;                       // N
    float* norm_dst = ws + N;                   // N
    _Float16* Wt16  = (_Float16*)(ws + 2L * N); // 128*128 halves = 8192 floats
    _Float16* hs    = (_Float16*)(ws + 2L * N + 8192);     // NH halves = NH/2 floats
    float* bufT     = ws + 2L * N + 8192 + NH / 2;          // NH
    float* bufA     = bufT + NH;                            // NH
    float* bufH     = bufA + NH;                            // NH
    (void)ws_size; (void)n_in;

    const int B = 256;

    // degrees -> norms
    gcn_zero_f32<<<cdiv(2L * N, B), B, 0, stream>>>(norm_src, 2L * N);
    gcn_degree<<<cdiv(E, B), B, 0, stream>>>(src, dst, norm_src, norm_dst, E);
    gcn_norm_finalize<<<cdiv(N, B), B, 0, stream>>>(norm_src, norm_dst, N);

    const float* h_cur = node_feats;
    for (int L = 0; L < 3; ++L) {
        gcn_wt_convert<<<cdiv(HDIM * HDIM, B), B, 0, stream>>>(Ws[L], Wt16);
        gcn_scale_f16<<<cdiv(NH, B), B, 0, stream>>>(h_cur, norm_src, hs, NH);
        gcn_gemm_wmma<<<cdiv(N, 128), B, 0, stream>>>(hs, Wt16, bufT, N);
        gcn_zero_f32<<<cdiv(NH, B), B, 0, stream>>>(bufA, NH);
        gcn_scatter<<<cdiv((long)(E + N) * 32, B), B, 0, stream>>>(src, dst, bufT, bufA, E, N);
        gcn_bias_relu<<<cdiv(NH, B), B, 0, stream>>>(bufA, norm_dst, bs[L], bufH, NH);
        h_cur = bufH;
    }

    // pooling into d_out [G, 256]
    gcn_zero_f32<<<cdiv(out_size, B), B, 0, stream>>>((float*)d_out, out_size);
    gcn_pool<<<cdiv((long)N * 32, B), B, 0, stream>>>(bufH, gid, W_pool, b_pool,
                                                      (float*)d_out, N);
}
